// Wide_BasicBlock_Q_35356170781050
// MI455X (gfx1250) — 8-bit matrix operand order repaired and checked by reading, compiled, not run on hardware
//
#include <hip/hip_runtime.h>
#include <math.h>

typedef __attribute__((ext_vector_type(8))) int v8i;

#define CCH 128
#define HWSP 256
#define NB 128
#define M_TOTAL (NB*HWSP)
#define KCHUNKS 18
#define BN_EPS 1e-5f

#define OFF_ACT1  0x0000000ull
#define OFF_ACT2  0x0400000ull
#define OFF_Y1    0x0800000ull
#define OFF_BP1   0x1800000ull
#define OFF_BP2   0x1828000ull
#define OFF_STATS 0x1850000ull

__global__ void bn1_stats_kernel(const float* __restrict__ x, float* __restrict__ stats) {
  __shared__ float ssum[256], ssq[256];
  int c = blockIdx.x, t = threadIdx.x;
  float s = 0.f, sq = 0.f;
  for (int i = t; i < M_TOTAL; i += 256) {
    int n = i >> 8, sp = i & 255;
    float v = x[(n * CCH + c) * HWSP + sp];
    s += v; sq += v * v;
  }
  ssum[t] = s; ssq[t] = sq; __syncthreads();
  for (int w = 128; w > 0; w >>= 1) {
    if (t < w) { ssum[t] += ssum[t + w]; ssq[t] += ssq[t + w]; }
    __syncthreads();
  }
  if (t == 0) {
    float mean = ssum[0] / (float)M_TOTAL;
    float var  = ssq[0] / (float)M_TOTAL - mean * mean;
    stats[c] = mean; stats[128 + c] = rsqrtf(var + BN_EPS);
  }
}

__global__ void quant_act1_kernel(const float* __restrict__ x,
                                  const float* __restrict__ gamma,
                                  const float* __restrict__ beta,
                                  const float* __restrict__ stats,
                                  unsigned char* __restrict__ act) {
  int idx = blockIdx.x * 256 + threadIdx.x;
  int c = idx & 127, m = idx >> 7;
  int n = m >> 8, sp = m & 255;
  float v = x[(n * CCH + c) * HWSP + sp];
  float h = (v - stats[c]) * stats[128 + c] * gamma[c] + beta[c];
  h = fminf(fmaxf(h, 0.f), 1.f);
  act[idx] = (unsigned char)(int)rintf(h * 15.f);
}

__global__ void maxtanh_kernel(const float* __restrict__ w1,
                               const float* __restrict__ w2,
                               float* __restrict__ stats) {
  __shared__ float sm[256];
  const float* w = blockIdx.x ? w2 : w1;
  int t = threadIdx.x;
  float m = 0.f;
  for (int i = t; i < CCH * CCH * 9; i += 256) m = fmaxf(m, fabsf(tanhf(w[i])));
  sm[t] = m; __syncthreads();
  for (int s = 128; s > 0; s >>= 1) {
    if (t < s) sm[t] = fmaxf(sm[t], sm[t + s]);
    __syncthreads();
  }
  if (t == 0) stats[512 + blockIdx.x] = sm[0];
}

__global__ void pack_w_kernel(const float* __restrict__ w1, const float* __restrict__ w2,
                              const float* __restrict__ pat1, const float* __restrict__ pat2,
                              const float* __restrict__ stats,
                              int* __restrict__ bp1, int* __restrict__ bp2) {
  int which = blockIdx.y;
  const float* w   = which ? w2   : w1;
  const float* pat = which ? pat2 : pat1;
  int* bp = which ? bp2 : bp1;
  float inv2m = 0.5f / stats[512 + which];
  int l = threadIdx.x;
  int nt = blockIdx.x / KCHUNKS, k = blockIdx.x % KCHUNKS;
  int tap = k >> 1, chalf = k & 1;
  int co = nt * 16 + (l & 15);
  float pm = pat[tap];
  for (int v = 0; v < 8; ++v) {
    unsigned int dw = 0;
    for (int bt = 0; bt < 4; ++bt) {
      int ci = chalf * 64 + (v >> 1) * 16 + ((l >= 16) ? 8 : 0) + (v & 1) * 4 + bt;
      float t = tanhf(w[(co * CCH + ci) * 9 + tap]);
      int q = (int)rintf((t * inv2m + 0.5f) * 15.f);
      int wi = 2 * q - 15;
      if (pm == 0.f) wi = 0;
      dw |= ((unsigned int)(wi & 0xff)) << (8 * bt);
    }
    bp[((nt * KCHUNKS + k) * 32 + l) * 8 + v] = (int)dw;
  }
}

__global__ void bn2_stats_kernel(const int* __restrict__ y1, float* __restrict__ stats) {
  __shared__ float ssum[256], ssq[256];
  int c = blockIdx.x, t = threadIdx.x;
  const float inv = 1.f / 225.f;
  float s = 0.f, sq = 0.f;
  for (int m = t; m < M_TOTAL; m += 256) {
    float v = (float)y1[m * CCH + c] * inv;
    s += v; sq += v * v;
  }
  ssum[t] = s; ssq[t] = sq; __syncthreads();
  for (int w = 128; w > 0; w >>= 1) {
    if (t < w) { ssum[t] += ssum[t + w]; ssq[t] += ssq[t + w]; }
    __syncthreads();
  }
  if (t == 0) {
    float mean = ssum[0] / (float)M_TOTAL;
    float var  = ssq[0] / (float)M_TOTAL - mean * mean;
    stats[256 + c] = mean; stats[384 + c] = rsqrtf(var + BN_EPS);
  }
}

__global__ void quant_act2_kernel(const int* __restrict__ y1,
                                  const float* __restrict__ gamma,
                                  const float* __restrict__ beta,
                                  const float* __restrict__ stats,
                                  unsigned char* __restrict__ act) {
  int idx = blockIdx.x * 256 + threadIdx.x;
  int c = idx & 127;
  float v = (float)y1[idx] * (1.f / 225.f);
  float h = (v - stats[256 + c]) * stats[384 + c] * gamma[c] + beta[c];
  h = fminf(fmaxf(h, 0.f), 1.f);
  act[idx] = (unsigned char)(int)rintf(h * 15.f);
}

__global__ void __launch_bounds__(256) conv_wmma_kernel(
    const unsigned char* __restrict__ act, const int* __restrict__ bpack,
    int* __restrict__ yout, float* __restrict__ fout, const float* __restrict__ xres,
    int mode) {
  int lane = threadIdx.x & 31;
  int wave = threadIdx.x >> 5;
  int mtile = blockIdx.x * 8 + wave;
  int wl = lane & 15;
  int hi = lane >> 4;
  int h = mtile & 15;
  int n = mtile >> 4;

  v8i z = {0, 0, 0, 0, 0, 0, 0, 0};
  v8i acc[8];
  for (int i = 0; i < 8; ++i) acc[i] = z;

  for (int k = 0; k < KCHUNKS; ++k) {
    int tap = k >> 1, chalf = k & 1;
    int dh = tap / 3 - 1, dw = tap % 3 - 1;
    int hh = h + dh, ww = wl + dw;
    bool inb = (hh >= 0) && (hh < 16) && (ww >= 0) && (ww < 16);
    v8i a = z;
    if (inb) {
      const unsigned char* base =
          act + ((((n << 4) + hh) << 4) + ww) * CCH + chalf * 64 + hi * 8;
      for (int j = 0; j < 4; ++j) {
        uint2 d = *(const uint2*)(base + j * 16);
        a[2 * j] = (int)d.x; a[2 * j + 1] = (int)d.y;
      }
    }
    const int* bb = bpack + (k * 32 + lane) * 8;
    __builtin_prefetch(bb + 32 * 8, 0, 1);
    for (int nt = 0; nt < 8; ++nt) {
      v8i b = *(const v8i*)(bb + nt * KCHUNKS * 32 * 8);
      acc[nt] = __builtin_amdgcn_wmma_i32_16x16x64_iu8(
           false, a,  true, b, acc[nt],
           false,  false);
    }
  }

  int mbase = mtile * 16;
  if (mode == 0) {
    for (int nt = 0; nt < 8; ++nt) {
      int co = nt * 16 + wl;
      for (int v = 0; v < 8; ++v) {
        int row = mbase + v + hi * 8;
        yout[row * CCH + co] = acc[nt][v];
      }
    }
  } else {
    const float inv = 1.f / 225.f;
    for (int nt = 0; nt < 8; ++nt) {
      int co = nt * 16 + wl;
      for (int v = 0; v < 8; ++v) {
        int w2 = v + hi * 8;
        int oidx = ((n * CCH + co) * 16 + h) * 16 + w2;
        fout[oidx] = (float)acc[nt][v] * inv + xres[oidx];
      }
    }
  }
}

extern "C" void kernel_launch(void* const* d_in, const int* in_sizes, int n_in,
                              void* d_out, int out_size, void* d_ws, size_t ws_size,
                              hipStream_t stream) {
  const float* x      = (const float*)d_in[0];
  const float* gamma1 = (const float*)d_in[1];
  const float* beta1  = (const float*)d_in[2];
  const float* gamma2 = (const float*)d_in[3];
  const float* beta2  = (const float*)d_in[4];
  const float* w1     = (const float*)d_in[5];
  const float* w2     = (const float*)d_in[6];
  const float* pat1   = (const float*)d_in[7];
  const float* pat2   = (const float*)d_in[8];
  float* out = (float*)d_out;

  char* ws = (char*)d_ws;
  unsigned char* act1 = (unsigned char*)(ws + OFF_ACT1);
  unsigned char* act2 = (unsigned char*)(ws + OFF_ACT2);
  int*   y1    = (int*)(ws + OFF_Y1);
  int*   bp1   = (int*)(ws + OFF_BP1);
  int*   bp2   = (int*)(ws + OFF_BP2);
  float* stats = (float*)(ws + OFF_STATS);

  bn1_stats_kernel<<<CCH, 256, 0, stream>>>(x, stats);
  quant_act1_kernel<<<(M_TOTAL * CCH) / 256, 256, 0, stream>>>(x, gamma1, beta1, stats, act1);

  maxtanh_kernel<<<2, 256, 0, stream>>>(w1, w2, stats);
  pack_w_kernel<<<dim3(8 * KCHUNKS, 2), 32, 0, stream>>>(w1, w2, pat1, pat2, stats, bp1, bp2);

  conv_wmma_kernel<<<M_TOTAL / 16 / 8, 256, 0, stream>>>(act1, bp1, y1, nullptr, nullptr, 0);

  bn2_stats_kernel<<<CCH, 256, 0, stream>>>(y1, stats);
  quant_act2_kernel<<<(M_TOTAL * CCH) / 256, 256, 0, stream>>>(y1, gamma2, beta2, stats, act2);

  conv_wmma_kernel<<<M_TOTAL / 16 / 8, 256, 0, stream>>>(act2, bp2, nullptr, out, x, 1);
}
